// MultiHeadAttention_20229295964402
// MI455X (gfx1250) — matrix operand order repaired and checked by reading, not compiled, not run on hardware
//
#include <hip/hip_runtime.h>
#include <hip/hip_bf16.h>
#include <math.h>

#define DEVINL __device__ __forceinline__

typedef __attribute__((ext_vector_type(16))) __bf16 v16bf;
typedef __attribute__((ext_vector_type(8)))  __bf16 v8bf;
typedef __attribute__((ext_vector_type(8)))  float  v8f;

namespace {
constexpr int Bb = 2, Ts = 2048, Cc = 1024, Hh = 16, HD = 64;
constexpr int M_ROWS = Bb * Ts;      // 4096 rows of X / O
constexpr int N_QKV  = 3 * Cc;       // 3072
}

union V16U { v16bf v; v8bf h[2]; };

DEVINL v8bf ldg8(const __bf16* p) { return *reinterpret_cast<const v8bf*>(p); }
DEVINL v16bf mk16(v8bf lo, v8bf hi) { V16U u; u.h[0] = lo; u.h[1] = hi; return u.v; }

DEVINL v8f wmma_bf16(v16bf a, v16bf b, v8f c) {
  // D(16x16,f32) = A(16x32,bf16) x B(32x16,bf16) + C
  return __builtin_amdgcn_wmma_f32_16x16x32_bf16(
      /*neg_a=*/false, a, /*neg_b=*/false, b,
      /*c_mod=*/(short)0, c, /*reuse_a=*/false, /*reuse_b=*/false);
}

// Row of a 16x16 C tile lives in one 16-lane half of the wave32: reduce there.
DEVINL float redmax16(float v) {
  v = fmaxf(v, __shfl_xor(v, 1, 32));
  v = fmaxf(v, __shfl_xor(v, 2, 32));
  v = fmaxf(v, __shfl_xor(v, 4, 32));
  v = fmaxf(v, __shfl_xor(v, 8, 32));
  return v;
}
DEVINL float redsum16(float v) {
  v += __shfl_xor(v, 1, 32);
  v += __shfl_xor(v, 2, 32);
  v += __shfl_xor(v, 4, 32);
  v += __shfl_xor(v, 8, 32);
  return v;
}

// ---------------- precision-convert kernels ----------------

__global__ void __launch_bounds__(256) cvt_bf16_kernel(const float* __restrict__ src,
                                                       __bf16* __restrict__ dst) {
  int i = blockIdx.x * 256 + threadIdx.x;
  dst[i] = (__bf16)src[i];
}

// dst[n*K + k] = bf16(src[k*N + n])  (store W^T so B-fragments are K-contiguous)
__global__ void __launch_bounds__(256) cvtT_bf16_kernel(const float* __restrict__ src,
                                                        __bf16* __restrict__ dst,
                                                        int K, int N) {
  int i = blockIdx.x * 256 + threadIdx.x;
  int n = i / K, k = i % K;
  dst[i] = (__bf16)src[k * N + n];
}

// ---------------- QKV GEMM: X(4096x1024) * Wattn(1024x3072) ----------------
// One wave per 32x64 output tile (2 A-frags x 4 B-frags -> 8 wmmas per k-step,
// 12 b128 loads per k-step = 1.5 loads/wmma). Each 64-wide N tile lies fully
// inside one head of one of K/Q/V, so the scatter indices are tile-uniform.

__global__ void __launch_bounds__(128) qkv_gemm_kernel(const __bf16* __restrict__ xb,
                                                       const __bf16* __restrict__ wT,
                                                       __bf16* __restrict__ Qb,
                                                       __bf16* __restrict__ Kb,
                                                       __bf16* __restrict__ VbT) {
  const int lane = threadIdx.x & 31;
  const int wid  = threadIdx.x >> 5;
  const int wgid = blockIdx.x * 4 + wid;
  const int ntn  = N_QKV / 64;                 // 48 tiles along N
  const int m0   = (wgid / ntn) * 32;
  const int n0   = (wgid % ntn) * 64;
  const int col  = lane & 15;
  const int hf   = lane >> 4;

  // A frag (16x32 bf16): lane row = col; elems i -> K = hf*16 + i (contiguous)
  const __bf16* aRow0 = xb + (size_t)(m0 + col) * Cc + hf * 16;
  const __bf16* aRow1 = aRow0 + (size_t)16 * Cc;
  // B frag (32x16 bf16) from W^T: lane col = col; elems i -> K = hf*16 + i (contiguous)
  const __bf16* bRow0 = wT + (size_t)(n0 + col) * Cc + hf * 16;

  v8f acc[2][4] = {};
  for (int kk = 0; kk < Cc; kk += 32) {
    const v16bf A0 = mk16(ldg8(aRow0 + kk), ldg8(aRow0 + kk + 8));
    const v16bf A1 = mk16(ldg8(aRow1 + kk), ldg8(aRow1 + kk + 8));
#pragma unroll
    for (int nb = 0; nb < 4; ++nb) {
      const __bf16* br = bRow0 + (size_t)nb * 16 * Cc + kk;
      const v16bf Bv = mk16(ldg8(br), ldg8(br + 8));
      acc[0][nb] = wmma_bf16(A0, Bv, acc[0][nb]);
      acc[1][nb] = wmma_bf16(A1, Bv, acc[1][nb]);
    }
  }

  // Reference split order: k = qkv[:, :C], q = [C:2C), v = [2C:3C)
  const int third = n0 / Cc;        // 0=k, 1=q, 2=v
  const int h     = (n0 % Cc) / HD; // 64-wide tile == exactly one head, d0 == 0

#pragma unroll
  for (int ma = 0; ma < 2; ++ma) {
#pragma unroll
    for (int nb = 0; nb < 4; ++nb) {
      const int d = nb * 16 + col;
#pragma unroll
      for (int r = 0; r < 8; ++r) {
        const int m = m0 + ma * 16 + r + 8 * hf;  // C layout: lanes16-31 hold rows r+8
        const int b = m / Ts, t = m % Ts;
        const float v = acc[ma][nb][r];
        if (third == 1) {                   // Q, fold in softmax scale 1/sqrt(64)
          Qb[((size_t)(b * Hh + h) * Ts + t) * HD + d] = (__bf16)(v * 0.125f);
        } else if (third == 0) {            // K, [B,H,T,HD]
          Kb[((size_t)(b * Hh + h) * Ts + t) * HD + d] = (__bf16)v;
        } else {                            // V stored transposed [B,H,HD,T]
          VbT[((size_t)(b * Hh + h) * HD + d) * Ts + t] = (__bf16)v;
        }
      }
    }
  }
}

// ---------------- Flash attention: per (b,h,q-tile of 16) ----------------

__global__ void __launch_bounds__(128) attn_kernel(const __bf16* __restrict__ Qb,
                                                   const __bf16* __restrict__ Kb,
                                                   const __bf16* __restrict__ VbT,
                                                   __bf16* __restrict__ Ob) {
  __shared__ __align__(16) __bf16 pLds[4][16][32];   // per-wave P transpose buffer

  const int lane = threadIdx.x & 31;
  const int wid  = threadIdx.x >> 5;
  const int w    = blockIdx.x * 4 + wid;
  const int nqt  = Ts / 16;                          // 128 q-tiles
  const int bh   = w / nqt;
  const int q0   = (w % nqt) * 16;
  const int col  = lane & 15;
  const int hf   = lane >> 4;

  const __bf16* Qp = Qb  + (size_t)bh * Ts * HD;
  const __bf16* Kp = Kb  + (size_t)bh * Ts * HD;
  const __bf16* Vp = VbT + (size_t)bh * HD * Ts;

  // Q fragment, 16x64 = two 16x32 A-frags, loaded once
  const __bf16* qBase = Qp + (size_t)(q0 + col) * HD + hf * 16;
  const v16bf Aq0 = mk16(ldg8(qBase),      ldg8(qBase + 8));
  const v16bf Aq1 = mk16(ldg8(qBase + 32), ldg8(qBase + 40));

  v8f o[4] = {v8f{}, v8f{}, v8f{}, v8f{}};
  float mrow[8], lrow[8];
#pragma unroll
  for (int r = 0; r < 8; ++r) { mrow[r] = -__builtin_inff(); lrow[r] = 0.0f; }

  for (int k0 = 0; k0 <= q0 + 15; k0 += 32) {        // causal: only needed key tiles
    // ---- S = Q K^T for 32 keys (two 16x16 C tiles), K-dim = HD = 64 ----
    const __bf16* k0p = Kp + (size_t)(k0 + col) * HD + hf * 16;   // keys k0..k0+15
    const __bf16* k1p = k0p + 16 * HD;                            // keys k0+16..k0+31
    v8f s0 = {}, s1 = {};
    s0 = wmma_bf16(Aq0, mk16(ldg8(k0p),      ldg8(k0p + 8)),  s0);
    s0 = wmma_bf16(Aq1, mk16(ldg8(k0p + 32), ldg8(k0p + 40)), s0);
    s1 = wmma_bf16(Aq0, mk16(ldg8(k1p),      ldg8(k1p + 8)),  s1);
    s1 = wmma_bf16(Aq1, mk16(ldg8(k1p + 32), ldg8(k1p + 40)), s1);

    // ---- causal mask + online softmax (row lives in a 16-lane half) ----
    const int kc0 = k0 + col, kc1 = k0 + 16 + col;
#pragma unroll
    for (int r = 0; r < 8; ++r) {
      const int qi = q0 + r + 8 * hf;
      const float a  = (kc0 <= qi) ? s0[r] : -__builtin_inff();
      const float b2 = (kc1 <= qi) ? s1[r] : -__builtin_inff();
      float mt = redmax16(fmaxf(a, b2));
      const float mn = fmaxf(mrow[r], mt);
      const float alpha = __expf(mrow[r] - mn);
      const float p0 = __expf(a - mn);
      const float p1 = __expf(b2 - mn);
      lrow[r] = lrow[r] * alpha + redsum16(p0 + p1);
      mrow[r] = mn;
#pragma unroll
      for (int t = 0; t < 4; ++t) o[t][r] *= alpha;
      const int row = r + 8 * hf;
      pLds[wid][row][col]      = (__bf16)p0;         // C-layout -> LDS
      pLds[wid][row][col + 16] = (__bf16)p1;
    }
    __builtin_amdgcn_wave_barrier();
    // read back in A-fragment layout: row = col, elems i -> K = hf*16 + i (contiguous)
    const __bf16* pptr = &pLds[wid][col][hf * 16];
    const v16bf Ap = mk16(*reinterpret_cast<const v8bf*>(pptr),
                          *reinterpret_cast<const v8bf*>(pptr + 8));
    __builtin_amdgcn_wave_barrier();

    // ---- O += P (16x32) * V (32x64), V^T gives contiguous B-fragments ----
#pragma unroll
    for (int t = 0; t < 4; ++t) {
      const __bf16* vp = Vp + (size_t)(t * 16 + col) * Ts + k0 + hf * 16;
      o[t] = wmma_bf16(Ap, mk16(ldg8(vp), ldg8(vp + 8)), o[t]);
    }
  }

  // ---- normalize and store O as bf16 [B*T, C] for the projection GEMM ----
  const int hh = bh % Hh, bb = bh / Hh;
#pragma unroll
  for (int r = 0; r < 8; ++r) {
    const int qi = q0 + r + 8 * hf;
    const float inv = 1.0f / lrow[r];
    __bf16* orow = Ob + (size_t)(bb * Ts + qi) * Cc + hh * HD + col;
    orow[0]  = (__bf16)(o[0][r] * inv);
    orow[16] = (__bf16)(o[1][r] * inv);
    orow[32] = (__bf16)(o[2][r] * inv);
    orow[48] = (__bf16)(o[3][r] * inv);
  }
}

// ---------------- Output projection: O(4096x1024) * Wproj(1024x1024) + b ----
// Same 32x64 register blocking as the QKV GEMM; fp32 result straight to d_out.

__global__ void __launch_bounds__(128) proj_gemm_kernel(const __bf16* __restrict__ Ob,
                                                        const __bf16* __restrict__ wpT,
                                                        const float* __restrict__ bias,
                                                        float* __restrict__ out) {
  const int lane = threadIdx.x & 31;
  const int wid  = threadIdx.x >> 5;
  const int wgid = blockIdx.x * 4 + wid;
  const int ntn  = Cc / 64;                    // 16 tiles along N
  const int m0   = (wgid / ntn) * 32;
  const int n0   = (wgid % ntn) * 64;
  const int col  = lane & 15;
  const int hf   = lane >> 4;

  const __bf16* aRow0 = Ob + (size_t)(m0 + col) * Cc + hf * 16;
  const __bf16* aRow1 = aRow0 + (size_t)16 * Cc;
  const __bf16* bRow0 = wpT + (size_t)(n0 + col) * Cc + hf * 16;

  v8f acc[2][4] = {};
  for (int kk = 0; kk < Cc; kk += 32) {
    const v16bf A0 = mk16(ldg8(aRow0 + kk), ldg8(aRow0 + kk + 8));
    const v16bf A1 = mk16(ldg8(aRow1 + kk), ldg8(aRow1 + kk + 8));
#pragma unroll
    for (int nb = 0; nb < 4; ++nb) {
      const __bf16* br = bRow0 + (size_t)nb * 16 * Cc + kk;
      const v16bf Bv = mk16(ldg8(br), ldg8(br + 8));
      acc[0][nb] = wmma_bf16(A0, Bv, acc[0][nb]);
      acc[1][nb] = wmma_bf16(A1, Bv, acc[1][nb]);
    }
  }

#pragma unroll
  for (int nb = 0; nb < 4; ++nb) {
    const float bn = bias[n0 + nb * 16 + col];
#pragma unroll
    for (int ma = 0; ma < 2; ++ma) {
#pragma unroll
      for (int r = 0; r < 8; ++r) {
        const int m = m0 + ma * 16 + r + 8 * hf;
        out[(size_t)m * Cc + n0 + nb * 16 + col] = acc[ma][nb][r] + bn;
      }
    }
  }
}

// ---------------- host launch ----------------

extern "C" void kernel_launch(void* const* d_in, const int* in_sizes, int n_in,
                              void* d_out, int out_size, void* d_ws, size_t ws_size,
                              hipStream_t stream) {
  const float* x      = (const float*)d_in[0];
  const float* w_attn = (const float*)d_in[1];
  const float* w_proj = (const float*)d_in[2];
  const float* b_proj = (const float*)d_in[3];
  // d_in[4] = mask: causal structure is computed inline; unused.
  float* out = (float*)d_out;

  char* ws = (char*)d_ws;
  size_t off = 0;
  auto wsalloc = [&](size_t bytes) -> char* {
    char* p = ws + off;
    off += (bytes + 255) & ~(size_t)255;
    return p;
  };
  const size_t headElems = (size_t)Bb * Hh * Ts * HD;   // 4.19M
  __bf16* xb  = (__bf16*)wsalloc((size_t)M_ROWS * Cc * 2);   // x in bf16
  __bf16* waT = (__bf16*)wsalloc((size_t)N_QKV * Cc * 2);    // w_attn^T bf16
  __bf16* wpT = (__bf16*)wsalloc((size_t)Cc * Cc * 2);       // w_proj^T bf16
  __bf16* Qb  = (__bf16*)wsalloc(headElems * 2);             // [B,H,T,HD], pre-scaled
  __bf16* Kb  = (__bf16*)wsalloc(headElems * 2);             // [B,H,T,HD]
  __bf16* VbT = (__bf16*)wsalloc(headElems * 2);             // [B,H,HD,T]
  __bf16* Ob  = (__bf16*)wsalloc((size_t)M_ROWS * Cc * 2);   // attention out bf16
  (void)ws_size; (void)in_sizes; (void)n_in; (void)out_size;

  cvt_bf16_kernel<<<(M_ROWS * Cc) / 256, 256, 0, stream>>>(x, xb);
  cvtT_bf16_kernel<<<(Cc * N_QKV) / 256, 256, 0, stream>>>(w_attn, waT, Cc, N_QKV);
  cvtT_bf16_kernel<<<(Cc * Cc) / 256, 256, 0, stream>>>(w_proj, wpT, Cc, Cc);

  qkv_gemm_kernel<<<(M_ROWS / 32) * (N_QKV / 64) / 4, 128, 0, stream>>>(xb, waT, Qb, Kb, VbT);
  attn_kernel<<<Bb * Hh * (Ts / 16) / 4, 128, 0, stream>>>(Qb, Kb, VbT, Ob);
  proj_gemm_kernel<<<(M_ROWS / 32) * (Cc / 64) / 4, 128, 0, stream>>>(Ob, wpT, b_proj, out);
}
